// simBNN_28037546509026
// MI455X (gfx1250) — 8-bit matrix operand order repaired and checked by reading, compiled, not run on hardware
//
#include <hip/hip_runtime.h>
#include <hip/hip_bf16.h>
#include <stdint.h>

typedef int v8i __attribute__((ext_vector_type(8)));

#define GAS __attribute__((address_space(1)))
typedef const signed char GAS* gcp;

__device__ __forceinline__ gcp to_gas(const signed char* p) {
    return (gcp)(unsigned long long)p;
}
__device__ __forceinline__ int ld_gi32(gcp p, int off) {
    return *(const GAS int*)(p + off);
}

__device__ __forceinline__ signed char fsign8(float x) {
    return x > 0.f ? (signed char)1 : (x < 0.f ? (signed char)-1 : (signed char)0);
}

__global__ void conv1_kernel(const float* __restrict__ x, const float* __restrict__ w,
                             const float* __restrict__ bias, float* __restrict__ out, int B) {
    const int OC = 64, OH = 55, OW = 55, IH = 224, IW = 224;
    long long tid = (long long)blockIdx.x * blockDim.x + threadIdx.x;
    long long total = (long long)B * OH * OW * OC;
    if (tid >= total) return;
    int oc = (int)(tid % OC); long long r = tid / OC;
    int ow = (int)(r % OW); r /= OW;
    int oh = (int)(r % OH); int b = (int)(r / OH);
    float acc = bias[oc];
    for (int ic = 0; ic < 3; ++ic) {
        const float* xp = x + ((long long)(b * 3 + ic) * IH) * IW;
        const float* wp = w + ((oc * 3 + ic) * 11) * 11;
        for (int kh = 0; kh < 11; ++kh) {
            int y = oh * 4 + kh - 2;
            if (y < 0 || y >= IH) continue;
            for (int kw = 0; kw < 11; ++kw) {
                int xx = ow * 4 + kw - 2;
                if (xx < 0 || xx >= IW) continue;
                acc = fmaf(xp[(long long)y * IW + xx], wp[kh * 11 + kw], acc);
            }
        }
    }
    out[tid] = acc;
}

__global__ void maxpool_nhwc(const float* __restrict__ in, float* __restrict__ out,
                             long long total, int C, int IH, int IW, int OH, int OW) {
    long long tid = (long long)blockIdx.x * blockDim.x + threadIdx.x;
    if (tid >= total) return;
    int c = (int)(tid % C); long long r = tid / C;
    int ow = (int)(r % OW); r /= OW;
    int oh = (int)(r % OH); int b = (int)(r / OH);
    float m = -INFINITY;
    for (int i = 0; i < 3; ++i)
        for (int j = 0; j < 3; ++j) {
            long long idx = (((long long)b * IH + (oh * 2 + i)) * IW + (ow * 2 + j)) * C + c;
            m = fmaxf(m, in[idx]);
        }
    out[tid] = m;
}

__global__ void binarize_k(const float* __restrict__ in, signed char* __restrict__ out, long long n) {
    long long tid = (long long)blockIdx.x * blockDim.x + threadIdx.x;
    if (tid >= n) return;
    out[tid] = fsign8(in[tid]);
}

__global__ void zero_bytes(signed char* __restrict__ p, int n) {
    int i = blockIdx.x * blockDim.x + threadIdx.x;
    if (i < n) p[i] = 0;
}

__global__ void pack_conv_w(const float* __restrict__ w, signed char* __restrict__ out,
                            int OC, int IC, int KH, int KW) {
    long long total = (long long)OC * IC * KH * KW;
    long long tid = (long long)blockIdx.x * blockDim.x + threadIdx.x;
    if (tid >= total) return;
    int ic = (int)(tid % IC); long long r = tid / IC;
    int kw = (int)(r % KW); r /= KW;
    int kh = (int)(r % KH); int oc = (int)(r / KH);
    out[tid] = fsign8(w[(((long long)oc * IC + ic) * KH + kh) * KW + kw]);
}

__global__ void pack_fc_w(const float* __restrict__ w, signed char* __restrict__ out,
                          int M, int Mpad, int K, int permute) {
    long long total = (long long)Mpad * K;
    long long tid = (long long)blockIdx.x * blockDim.x + threadIdx.x;
    if (tid >= total) return;
    int m = (int)(tid / K);
    int k = (int)(tid % K);
    if (m >= M) { out[tid] = 0; return; }
    int ksrc = k;
    if (permute) { int s = k >> 8; int c = k & 255; ksrc = c * 36 + s; }
    out[tid] = fsign8(w[(long long)m * K + ksrc]);
}

template <int IC, int KH, int KW, int PAD>
__global__ void bconv_iu8(const signed char* __restrict__ Aw,
                          const signed char* __restrict__ act,
                          const signed char* __restrict__ zbuf,
                          const float* __restrict__ bias,
                          float* __restrict__ out,
                          int Mtot, int IH, int IW, int OH, int OW, int Ncols) {
    const int K = KH * KW * IC;
    int wave = (int)((blockIdx.x * blockDim.x + threadIdx.x) >> 5);
    int mtiles = Mtot >> 4;
    int ntiles = Ncols >> 6;
    if (wave >= mtiles * ntiles) return;
    int mtile = wave % mtiles;
    int nt = wave / mtiles;
    int lane = threadIdx.x & 31;
    int half = lane >> 4;
    int l16 = lane & 15;

    gcp Ar = to_gas(Aw) + (long long)((mtile << 4) + l16) * K + (half << 4);
    gcp zb = to_gas(zbuf) + (half << 4);

    int ncol[4], oh[4], ow[4];
    gcp pbase[4];
#pragma unroll
    for (int t = 0; t < 4; ++t) {
        int n = (nt << 6) + (t << 4) + l16;
        ncol[t] = n;
        int r = n; ow[t] = r % OW; r /= OW; oh[t] = r % OH; int bb = r / OH;
        pbase[t] = to_gas(act) + (long long)bb * IH * IW * IC + (half << 4);
    }

    v8i acc[4];
#pragma unroll
    for (int t = 0; t < 4; ++t) acc[t] = (v8i){0, 0, 0, 0, 0, 0, 0, 0};

    for (int kh = 0; kh < KH; ++kh) {
        int yrow[4]; bool okr[4];
#pragma unroll
        for (int t = 0; t < 4; ++t) {
            int y = oh[t] + kh - PAD;
            okr[t] = (unsigned)y < (unsigned)IH;
            yrow[t] = y * IW;
        }
        gcp ap = Ar + kh * (KW * IC);
#pragma unroll
        for (int kw = 0; kw < KW; ++kw) {
            gcp bp[4];
#pragma unroll
            for (int t = 0; t < 4; ++t) {
                int xx = ow[t] + kw - PAD;
                bool ok = okr[t] && ((unsigned)xx < (unsigned)IW);
                bp[t] = ok ? (pbase[t] + (long long)(yrow[t] + xx) * IC) : zb;
            }
#pragma unroll
            for (int icb = 0; icb < IC; icb += 64) {
                v8i a;
#pragma unroll
                for (int v = 0; v < 8; ++v)
                    a[v] = ld_gi32(ap, kw * IC + icb + ((v >> 2) << 5) + ((v & 3) << 2));
#pragma unroll
                for (int t = 0; t < 4; ++t) {
                    v8i bm;
#pragma unroll
                    for (int v = 0; v < 8; ++v)
                        bm[v] = ld_gi32(bp[t], icb + ((v >> 2) << 5) + ((v & 3) << 2));
                    acc[t] = __builtin_amdgcn_wmma_i32_16x16x64_iu8(true, a, true, bm, acc[t],
                                                                    false, false);
                }
            }
        }
    }

#pragma unroll
    for (int r = 0; r < 8; ++r) {
        int mo = (mtile << 4) + r + (half << 3);
        float bv = bias[mo];
#pragma unroll
        for (int t = 0; t < 4; ++t)
            out[(long long)ncol[t] * Mtot + mo] = (float)acc[t][r] + bv;
    }
}

__global__ void bfc_iu8(const signed char* __restrict__ Aw,
                        const signed char* __restrict__ act,
                        float* __restrict__ out,
                        int Mreal, int Mpad, int N, int K) {
    int wave = (int)((blockIdx.x * blockDim.x + threadIdx.x) >> 5);
    int mtiles = Mpad >> 4;
    int ntiles = N >> 6;
    if (wave >= mtiles * ntiles) return;
    int mtile = wave % mtiles;
    int nt = wave / mtiles;
    int lane = threadIdx.x & 31;
    int half = lane >> 4;
    int l16 = lane & 15;

    gcp ap = to_gas(Aw) + (long long)((mtile << 4) + l16) * K + (half << 4);
    gcp bp[4];
#pragma unroll
    for (int t = 0; t < 4; ++t) {
        int n = (nt << 6) + (t << 4) + l16;
        bp[t] = to_gas(act) + (long long)n * K + (half << 4);
    }

    v8i acc[4];
#pragma unroll
    for (int t = 0; t < 4; ++t) acc[t] = (v8i){0, 0, 0, 0, 0, 0, 0, 0};

    for (int k0 = 0; k0 < K; k0 += 64) {
        v8i a;
#pragma unroll
        for (int v = 0; v < 8; ++v)
            a[v] = ld_gi32(ap, ((v >> 2) << 5) + ((v & 3) << 2));
#pragma unroll
        for (int t = 0; t < 4; ++t) {
            v8i bm;
#pragma unroll
            for (int v = 0; v < 8; ++v)
                bm[v] = ld_gi32(bp[t], ((v >> 2) << 5) + ((v & 3) << 2));
            acc[t] = __builtin_amdgcn_wmma_i32_16x16x64_iu8(true, a, true, bm, acc[t],
                                                            false, false);
        }
        ap += 64;
#pragma unroll
        for (int t = 0; t < 4; ++t) bp[t] += 64;
    }

#pragma unroll
    for (int r = 0; r < 8; ++r) {
        int mo = (mtile << 4) + r + (half << 3);
        if (mo >= Mreal) continue;
#pragma unroll
        for (int t = 0; t < 4; ++t) {
            int n = (nt << 6) + (t << 4) + l16;
            out[(long long)n * Mreal + mo] = (float)acc[t][r];
        }
    }
}

static inline long long cdiv_ll(long long a, long long b) { return (a + b - 1) / b; }

extern "C" void kernel_launch(void* const* d_in, const int* in_sizes, int n_in,
                              void* d_out, int out_size, void* d_ws, size_t ws_size,
                              hipStream_t stream) {
    (void)in_sizes; (void)n_in; (void)out_size; (void)ws_size;
    const float* x   = (const float*)d_in[0];
    const float* w1  = (const float*)d_in[1];
    const float* b1  = (const float*)d_in[2];
    const float* w2  = (const float*)d_in[3];
    const float* b2  = (const float*)d_in[4];
    const float* w3  = (const float*)d_in[5];
    const float* b3  = (const float*)d_in[6];
    const float* w4  = (const float*)d_in[7];
    const float* b4  = (const float*)d_in[8];
    const float* w5  = (const float*)d_in[9];
    const float* b5  = (const float*)d_in[10];
    const float* fw1 = (const float*)d_in[11];
    const float* fw2 = (const float*)d_in[12];
    const float* fw3 = (const float*)d_in[13];
    float* out = (float*)d_out;

    const int B = 128;
    char* base = (char*)d_ws;
    size_t cur = 0;
    auto alloc = [&](size_t bytes) -> void* {
        cur = (cur + 255) & ~(size_t)255;
        void* p = base + cur;
        cur += bytes;
        return p;
    };

    long long n_h1  = (long long)B * 55 * 55 * 64;
    long long n_h1p = (long long)B * 27 * 27 * 64;
    long long n_h2  = (long long)B * 27 * 27 * 192;
    long long n_h2p = (long long)B * 13 * 13 * 192;
    long long n_h3  = (long long)B * 13 * 13 * 384;
    long long n_h4  = (long long)B * 13 * 13 * 256;
    long long n_h5  = n_h4;
    long long n_h5p = (long long)B * 6 * 6 * 256;

    signed char* zbuf = (signed char*)alloc(1024);
    float* h1  = (float*)alloc(n_h1 * 4);
    float* h1p = (float*)alloc(n_h1p * 4);
    signed char* s1 = (signed char*)alloc(n_h1p);
    signed char* w2s = (signed char*)alloc((size_t)192 * 5 * 5 * 64);
    float* h2  = (float*)alloc(n_h2 * 4);
    float* h2p = (float*)alloc(n_h2p * 4);
    signed char* s2 = (signed char*)alloc(n_h2p);
    signed char* w3s = (signed char*)alloc((size_t)384 * 3 * 3 * 192);
    float* h3  = (float*)alloc(n_h3 * 4);
    signed char* s3 = (signed char*)alloc(n_h3);
    signed char* w4s = (signed char*)alloc((size_t)256 * 3 * 3 * 384);
    float* h4  = (float*)alloc(n_h4 * 4);
    signed char* s4 = (signed char*)alloc(n_h4);
    signed char* w5s = (signed char*)alloc((size_t)256 * 3 * 3 * 256);
    float* h5  = (float*)alloc(n_h5 * 4);
    float* h5p = (float*)alloc(n_h5p * 4);
    signed char* s5 = (signed char*)alloc(n_h5p);
    signed char* fw1s = (signed char*)alloc((size_t)4096 * 9216);
    float* a1 = (float*)alloc((size_t)B * 4096 * 4);
    signed char* sa1 = (signed char*)alloc((size_t)B * 4096);
    signed char* fw2s = (signed char*)alloc((size_t)1024 * 4096);
    float* a2 = (float*)alloc((size_t)B * 1024 * 4);
    signed char* sa2 = (signed char*)alloc((size_t)B * 1024);
    signed char* fw3s = (signed char*)alloc((size_t)1008 * 1024);

    const int T = 256;

    zero_bytes<<<4, 256, 0, stream>>>(zbuf, 1024);

    conv1_kernel<<<(int)cdiv_ll(n_h1, T), T, 0, stream>>>(x, w1, b1, h1, B);
    maxpool_nhwc<<<(int)cdiv_ll(n_h1p, T), T, 0, stream>>>(h1, h1p, n_h1p, 64, 55, 55, 27, 27);
    binarize_k<<<(int)cdiv_ll(n_h1p, T), T, 0, stream>>>(h1p, s1, n_h1p);

    pack_conv_w<<<(int)cdiv_ll((long long)192 * 64 * 25, T), T, 0, stream>>>(w2, w2s, 192, 64, 5, 5);
    {
        int Ncols = B * 27 * 27;
        long long waves = (192 / 16) * ((long long)Ncols / 64);
        bconv_iu8<64, 5, 5, 2><<<(int)cdiv_ll(waves * 32, T), T, 0, stream>>>(
            w2s, s1, zbuf, b2, h2, 192, 27, 27, 27, 27, Ncols);
    }
    maxpool_nhwc<<<(int)cdiv_ll(n_h2p, T), T, 0, stream>>>(h2, h2p, n_h2p, 192, 27, 27, 13, 13);
    binarize_k<<<(int)cdiv_ll(n_h2p, T), T, 0, stream>>>(h2p, s2, n_h2p);

    pack_conv_w<<<(int)cdiv_ll((long long)384 * 192 * 9, T), T, 0, stream>>>(w3, w3s, 384, 192, 3, 3);
    {
        int Ncols = B * 13 * 13;
        long long waves = (384 / 16) * ((long long)Ncols / 64);
        bconv_iu8<192, 3, 3, 1><<<(int)cdiv_ll(waves * 32, T), T, 0, stream>>>(
            w3s, s2, zbuf, b3, h3, 384, 13, 13, 13, 13, Ncols);
    }
    binarize_k<<<(int)cdiv_ll(n_h3, T), T, 0, stream>>>(h3, s3, n_h3);

    pack_conv_w<<<(int)cdiv_ll((long long)256 * 384 * 9, T), T, 0, stream>>>(w4, w4s, 256, 384, 3, 3);
    {
        int Ncols = B * 13 * 13;
        long long waves = (256 / 16) * ((long long)Ncols / 64);
        bconv_iu8<384, 3, 3, 1><<<(int)cdiv_ll(waves * 32, T), T, 0, stream>>>(
            w4s, s3, zbuf, b4, h4, 256, 13, 13, 13, 13, Ncols);
    }
    binarize_k<<<(int)cdiv_ll(n_h4, T), T, 0, stream>>>(h4, s4, n_h4);

    pack_conv_w<<<(int)cdiv_ll((long long)256 * 256 * 9, T), T, 0, stream>>>(w5, w5s, 256, 256, 3, 3);
    {
        int Ncols = B * 13 * 13;
        long long waves = (256 / 16) * ((long long)Ncols / 64);
        bconv_iu8<256, 3, 3, 1><<<(int)cdiv_ll(waves * 32, T), T, 0, stream>>>(
            w5s, s4, zbuf, b5, h5, 256, 13, 13, 13, 13, Ncols);
    }
    maxpool_nhwc<<<(int)cdiv_ll(n_h5p, T), T, 0, stream>>>(h5, h5p, n_h5p, 256, 13, 13, 6, 6);
    binarize_k<<<(int)cdiv_ll(n_h5p, T), T, 0, stream>>>(h5p, s5, n_h5p);

    pack_fc_w<<<(int)cdiv_ll((long long)4096 * 9216, T), T, 0, stream>>>(fw1, fw1s, 4096, 4096, 9216, 1);
    {
        long long waves = (4096 / 16) * (B / 64);
        bfc_iu8<<<(int)cdiv_ll(waves * 32, T), T, 0, stream>>>(fw1s, s5, a1, 4096, 4096, B, 9216);
    }
    binarize_k<<<(int)cdiv_ll((long long)B * 4096, T), T, 0, stream>>>(a1, sa1, (long long)B * 4096);

    pack_fc_w<<<(int)cdiv_ll((long long)1024 * 4096, T), T, 0, stream>>>(fw2, fw2s, 1024, 1024, 4096, 0);
    {
        long long waves = (1024 / 16) * (B / 64);
        bfc_iu8<<<(int)cdiv_ll(waves * 32, T), T, 0, stream>>>(fw2s, sa1, a2, 1024, 1024, B, 4096);
    }
    binarize_k<<<(int)cdiv_ll((long long)B * 1024, T), T, 0, stream>>>(a2, sa2, (long long)B * 1024);

    pack_fc_w<<<(int)cdiv_ll((long long)1008 * 1024, T), T, 0, stream>>>(fw3, fw3s, 1000, 1008, 1024, 0);
    {
        long long waves = (1008 / 16) * (B / 64);
        bfc_iu8<<<(int)cdiv_ll(waves * 32, T), T, 0, stream>>>(fw3s, sa2, out, 1000, 1008, B, 1024);
    }
}
